// GCN2_29231547416621
// MI455X (gfx1250) — matrix operand order repaired and checked by reading, not compiled, not run on hardware
//
#include <hip/hip_runtime.h>
#include <math.h>
#include <stdint.h>

typedef __attribute__((ext_vector_type(16))) __bf16 v16bf;
typedef __attribute__((ext_vector_type(8)))  float  v8f;
typedef __attribute__((ext_vector_type(4)))  int    v4i;

#define NNODES 100000
#define NEDGES 3200000
#define FIN    512
#define HDIM   256
#define FOUT   64
#define NLAYER 8
#define ALPHA_ 0.1f
#define THETA_ 0.5f

// -------------------------------------------------- async global->LDS staging
#if defined(__has_builtin)
#if __has_builtin(__builtin_amdgcn_global_load_async_to_lds_b128)
#define HAVE_ASYNC_LDS 1
#endif
#endif
#ifndef HAVE_ASYNC_LDS
#define HAVE_ASYNC_LDS 0
#endif

#if HAVE_ASYNC_LDS
typedef v4i __attribute__((address_space(1))) gv4i;   // global 128-bit vector
typedef v4i __attribute__((address_space(3))) lv4i;   // LDS 128-bit vector

__device__ inline void async_cp16(const void* g, void* l) {
  // AS1 global pointers are numerically identical to generic pointers;
  // generic LDS pointers carry the LDS byte offset in their low 32 bits
  // (ISA 10.2 aperture mapping), so integer casts produce valid AS pointers.
  __builtin_amdgcn_global_load_async_to_lds_b128(
      (gv4i*)(uintptr_t)g,
      (lv4i*)(unsigned int)(uintptr_t)l,
      0, 0);
}
__device__ inline void wait_async0() {
#if __has_builtin(__builtin_amdgcn_s_wait_asynccnt)
  __builtin_amdgcn_s_wait_asynccnt(0);
#else
  asm volatile("s_wait_asynccnt 0x0" ::: "memory");
#endif
}
#endif

// ---------------------------------------------------------------- WMMA helper
__device__ inline v8f wmma_bf16(v16bf a, v16bf b, v8f c) {
  // (neg_a, A, neg_b, B, c_mod, C, reuse_a, reuse_b)
  return __builtin_amdgcn_wmma_f32_16x16x32_bf16(false, a, false, b, (short)0, c,
                                                 false, false);
}

// ---------------------------------------------------------------- small utils
__global__ void fill_f32(float* p, float v, int n) {
  int i = blockIdx.x * blockDim.x + threadIdx.x;
  if (i < n) p[i] = v;
}

__global__ void degree_kernel(const int* __restrict__ dst, float* __restrict__ deg, int E) {
  int e = blockIdx.x * blockDim.x + threadIdx.x;
  if (e < E) atomicAdd(&deg[dst[e]], 1.0f);
}

__global__ void rsqrt_kernel(const float* __restrict__ deg, float* __restrict__ dinv, int n) {
  int i = blockIdx.x * blockDim.x + threadIdx.x;
  if (i < n) {
    float d = deg[i];
    dinv[i] = (d > 0.0f) ? rsqrtf(d) : 0.0f;
  }
}

// Pack fp32 W[K,N] row-major into bf16 hi/lo WMMA-B layout:
// element index = ((kt*numNt + nt)*32 + lane)*16 + e
// lane<16 : N = nt*16+lane,    K = kt*32 + e
// lane>=16: N = nt*16+lane-16, K = kt*32 + 16 + e
__global__ void pack_weights(const float* __restrict__ W, __bf16* __restrict__ ph,
                             __bf16* __restrict__ pl, int K, int N) {
  int tid = blockIdx.x * blockDim.x + threadIdx.x;
  if (tid >= K * N) return;
  int e    = tid & 15;
  int lane = (tid >> 4) & 31;
  int rest = tid >> 9;           // kt*numNt + nt
  int numNt = N >> 4;
  int nt = rest % numNt;
  int kt = rest / numNt;
  int sel = lane >> 4;
  int n = nt * 16 + (lane & 15);
  int k = kt * 32 + sel * 16 + e;
  float f  = W[(size_t)k * N + n];
  float hf = __uint_as_float(__float_as_uint(f) & 0xFFFF0000u);
  ph[tid] = (__bf16)hf;
  pl[tid] = (__bf16)(f - hf);
}

// ---------------------------------------------------------------- WMMA GEMM
// C[M,N] = epilogue( A[M,K] @ B[K,N] ), B pre-packed bf16 hi/lo.
// Block = 8 waves sharing one 64-column group; the block's B panel (hi+lo,
// full K) is async-staged into LDS once, then each wave computes a 16x64 tile
// reading B from LDS. A fp32 rows are split on the fly into bf16 hi/lo;
// 3-term product (ah*bh + ah*bl + al*bh) recovers ~fp32 accuracy.
// epilogue: v = c2*acc + (Zres? c1*Zres[m,n] : 0) + (bias? bias[n] : 0); relu opt.
__global__ __launch_bounds__(256) void gemm_wmma(
    const float* __restrict__ A, const __bf16* __restrict__ Bh,
    const __bf16* __restrict__ Bl, float* __restrict__ C,
    const float* __restrict__ Zres, const float* __restrict__ bias,
    float c1, float c2, int doRelu, int M, int K, int N) {
  const int lane   = threadIdx.x & 31;
  const int wIb    = threadIdx.x >> 5;     // wave in block (0..7)
  const int numNG  = N >> 6;               // 64-wide column groups
  const int numMt  = M >> 4;
  const int nG     = blockIdx.x % numNG;
  const int mChunk = blockIdx.x / numNG;
  const int mTile  = mChunk * 8 + wIb;
  const int kSteps = K >> 5;
  const int numNt  = N >> 4;
  const int nt0    = nG << 2;

  extern __shared__ __align__(32) char gemm_smem[];
  __bf16* ldsH = (__bf16*)gemm_smem;
  __bf16* ldsL = ldsH + (size_t)kSteps * 2048;   // kSteps*4 tiles * 512 elems

  // ---- cooperative staging of this block's B panel (K x 64 cols, hi+lo)
  {
    const int units = kSteps * 256;              // 16-byte units per panel
    for (int lu = threadIdx.x; lu < units; lu += 256) {
      int half  = lu & 1;
      int lane2 = (lu >> 1) & 31;
      int tt    = (lu >> 6) & 3;
      int ks    = lu >> 8;
      size_t gu = (((size_t)(ks * numNt + nt0 + tt)) * 32 + lane2) * 2 + half;
#if HAVE_ASYNC_LDS
      async_cp16(Bh + gu * 8, ldsH + (size_t)lu * 8);
      async_cp16(Bl + gu * 8, ldsL + (size_t)lu * 8);
#else
      ((float4*)ldsH)[lu] = ((const float4*)Bh)[gu];
      ((float4*)ldsL)[lu] = ((const float4*)Bl)[gu];
#endif
    }
#if HAVE_ASYNC_LDS
    wait_async0();
#endif
    __syncthreads();
  }

  if (mTile >= numMt) return;

  const int mBase = mTile << 4;
  const int sel   = lane >> 4;
  const int l15   = lane & 15;
  const int row   = mBase + l15;

  v8f acc[4] = {};

  const float* arow = A + (size_t)row * K;
  const v16bf* BhV = (const v16bf*)ldsH;
  const v16bf* BlV = (const v16bf*)ldsL;
  for (int ks = 0; ks < kSteps; ++ks) {
    const int kk = ks << 5;
    // A lane layout used here (contiguous k, same as B pack): lanes<16: K kk+0..15,
    //                                 lanes>=16: K kk+16..31
    const float4* p0 = (const float4*)(arow + kk + sel * 16);
    const float4* p1 = (const float4*)(arow + kk + sel * 16 + 8);
    float4 f0 = p0[0], f1 = p0[1], f2 = p1[0], f3 = p1[1];
    float tmp[16] = {f0.x, f0.y, f0.z, f0.w, f1.x, f1.y, f1.z, f1.w,
                     f2.x, f2.y, f2.z, f2.w, f3.x, f3.y, f3.z, f3.w};
    v16bf ah, al;
#pragma unroll
    for (int i = 0; i < 16; ++i) {
      float f  = tmp[i];
      float hf = __uint_as_float(__float_as_uint(f) & 0xFFFF0000u);
      ah[i] = (__bf16)hf;
      al[i] = (__bf16)(f - hf);
    }
    const int bbase = (ks << 2) * 32 + lane;
#pragma unroll
    for (int t = 0; t < 4; ++t) {
      v16bf bh = BhV[bbase + t * 32];
      v16bf bl = BlV[bbase + t * 32];
      acc[t] = wmma_bf16(ah, bh, acc[t]);   // hi*hi
      acc[t] = wmma_bf16(ah, bl, acc[t]);   // hi*lo
      acc[t] = wmma_bf16(al, bh, acc[t]);   // lo*hi
    }
  }

  // C/D layout: lane<16: (M=r, N=lane) ; lane>=16: (M=8+r, N=lane-16)
#pragma unroll
  for (int r = 0; r < 8; ++r) {
    const int m = mBase + sel * 8 + r;
    const size_t rowOff = (size_t)m * N;
#pragma unroll
    for (int t = 0; t < 4; ++t) {
      const int n = (nG << 6) + (t << 4) + l15;
      float v = c2 * acc[t][r];
      if (Zres) v += c1 * Zres[rowOff + n];
      if (bias) v += bias[n];
      if (doRelu) v = fmaxf(v, 0.0f);
      C[rowOff + n] = v;
    }
  }
}

// ---------------------------------------------------------------- propagation
// agg initialized with the self-loop contribution: agg[i,:] = dinv[i]^2 * h[i,:]
__global__ void agg_init(const float* __restrict__ dinv, const float* __restrict__ h,
                         float* __restrict__ agg) {
  size_t i = (size_t)blockIdx.x * blockDim.x + threadIdx.x;   // float4 index
  if (i >= (size_t)NNODES * (HDIM / 4)) return;
  int node = (int)(i >> 6);                                   // 64 float4 per row
  float s = dinv[node];
  s *= s;
  float4 hv = ((const float4*)h)[i];
  float4 o  = {s * hv.x, s * hv.y, s * hv.z, s * hv.w};
  ((float4*)agg)[i] = o;
}

// edge-parallel SpMM: agg[dst,:] += dinv[src]*dinv[dst] * h[src,:]
// 64 lanes per edge, float4 per lane (256 features); h is L2-resident (102MB < 192MB L2)
__global__ __launch_bounds__(256) void spmm_kernel(
    const int* __restrict__ src, const int* __restrict__ dst,
    const float* __restrict__ dinv, const float* __restrict__ h,
    float* __restrict__ agg, int E) {
  int e = blockIdx.x * (blockDim.x / 64) + (threadIdx.x / 64);
  int t = threadIdx.x & 63;
  if (e >= E) return;
  int s = src[e], d = dst[e];
  float nrm = dinv[s] * dinv[d];
  float4 hv = ((const float4*)(h + (size_t)s * HDIM))[t];
  float* out = agg + (size_t)d * HDIM + t * 4;
  atomicAdd(out + 0, nrm * hv.x);
  atomicAdd(out + 1, nrm * hv.y);
  atomicAdd(out + 2, nrm * hv.z);
  atomicAdd(out + 3, nrm * hv.w);
}

__global__ void z_kernel(const float* __restrict__ agg, const float* __restrict__ x0,
                         float* __restrict__ z) {
  size_t i = (size_t)blockIdx.x * blockDim.x + threadIdx.x;   // float4 index
  if (i >= (size_t)NNODES * (HDIM / 4)) return;
  float4 a = ((const float4*)agg)[i];
  float4 x = ((const float4*)x0)[i];
  float4 o = {(1.0f - ALPHA_) * a.x + ALPHA_ * x.x,
              (1.0f - ALPHA_) * a.y + ALPHA_ * x.y,
              (1.0f - ALPHA_) * a.z + ALPHA_ * x.z,
              (1.0f - ALPHA_) * a.w + ALPHA_ * x.w};
  ((float4*)z)[i] = o;
}

// ---------------------------------------------------------------- softmax/argmax
// one wave (32 lanes) per row of 64 logits; wave32 shuffle reductions
__global__ __launch_bounds__(256) void softmax_argmax(
    const float* __restrict__ logits, float* __restrict__ soft,
    float* __restrict__ hard, int M) {
  int wave = (int)((blockIdx.x * blockDim.x + threadIdx.x) >> 5);
  int lane = threadIdx.x & 31;
  if (wave >= M) return;
  const float* row = logits + (size_t)wave * FOUT;
  float v0 = row[lane], v1 = row[lane + 32];
  float mv  = (v1 > v0) ? v1 : v0;
  int   idx = (v1 > v0) ? lane + 32 : lane;   // first-occurrence tie break below
#pragma unroll
  for (int off = 16; off > 0; off >>= 1) {
    float om = __shfl_xor(mv, off, 32);
    int   oi = __shfl_xor(idx, off, 32);
    if (om > mv || (om == mv && oi < idx)) { mv = om; idx = oi; }
  }
  float e0 = __expf(v0 - mv), e1 = __expf(v1 - mv);
  float s = e0 + e1;
#pragma unroll
  for (int off = 16; off > 0; off >>= 1) s += __shfl_xor(s, off, 32);
  float inv = 1.0f / s;
  soft[(size_t)wave * FOUT + lane]      = e0 * inv;
  soft[(size_t)wave * FOUT + lane + 32] = e1 * inv;
  if (lane == 0) hard[wave] = (float)idx;
}

// ---------------------------------------------------------------- launcher
extern "C" void kernel_launch(void* const* d_in, const int* in_sizes, int n_in,
                              void* d_out, int out_size, void* d_ws, size_t ws_size,
                              hipStream_t stream) {
  const float* x  = (const float*)d_in[0];
  const int*   ei = (const int*)d_in[1];          // [2, E] flat
  const float* W0 = (const float*)d_in[2];
  const float* b0 = (const float*)d_in[3];
  const float* Ws = (const float*)d_in[4];        // [L, H, H]
  const float* Wc = (const float*)d_in[5];
  const float* bc = (const float*)d_in[6];
  const int* srcE = ei;
  const int* dstE = ei + NEDGES;

  // --- workspace carve
  char* ws = (char*)d_ws;
  size_t off = 0;
  auto alloc = [&](size_t bytes) -> void* {
    void* p = ws + off;
    off = (off + bytes + 255) & ~(size_t)255;
    return p;
  };
  float* deg   = (float*)alloc((size_t)NNODES * 4);
  float* dinv  = (float*)alloc((size_t)NNODES * 4);
  float* x0    = (float*)alloc((size_t)NNODES * HDIM * 4);
  float* hA    = (float*)alloc((size_t)NNODES * HDIM * 4);
  float* agg   = (float*)alloc((size_t)NNODES * HDIM * 4);
  float* zb    = (float*)alloc((size_t)NNODES * HDIM * 4);
  __bf16* pW0h = (__bf16*)alloc((size_t)FIN * HDIM * 2);
  __bf16* pW0l = (__bf16*)alloc((size_t)FIN * HDIM * 2);
  __bf16* pWsh = (__bf16*)alloc((size_t)NLAYER * HDIM * HDIM * 2);
  __bf16* pWsl = (__bf16*)alloc((size_t)NLAYER * HDIM * HDIM * 2);
  __bf16* pWch = (__bf16*)alloc((size_t)HDIM * FOUT * 2);
  __bf16* pWcl = (__bf16*)alloc((size_t)HDIM * FOUT * 2);

  // --- output carve (return order: logits, embedding, soft_label, hard_label)
  float* outLogits = (float*)d_out;
  float* outEmb    = outLogits + (size_t)NNODES * FOUT;
  float* outSoft   = outEmb + (size_t)NNODES * HDIM;
  float* outHard   = outSoft + (size_t)NNODES * FOUT;

  const int TB = 256;

  auto launch_gemm = [&](const float* A, const __bf16* BhP, const __bf16* BlP,
                         float* C, const float* Zr, const float* bs,
                         float c1, float c2, int relu, int M, int K, int N) {
    int numNG  = N / 64;
    int numMt  = M / 16;
    int blocks = ((numMt + 7) / 8) * numNG;
    size_t smem = (size_t)(K / 32) * 8192;   // hi+lo B panel for 64 columns
    gemm_wmma<<<blocks, TB, smem, stream>>>(A, BhP, BlP, C, Zr, bs,
                                            c1, c2, relu, M, K, N);
  };

  // 1) degrees (self loop folded: deg starts at 1)
  fill_f32<<<(NNODES + TB - 1) / TB, TB, 0, stream>>>(deg, 1.0f, NNODES);
  degree_kernel<<<(NEDGES + TB - 1) / TB, TB, 0, stream>>>(dstE, deg, NEDGES);
  rsqrt_kernel<<<(NNODES + TB - 1) / TB, TB, 0, stream>>>(deg, dinv, NNODES);

  // 2) pack weights into WMMA-B bf16 hi/lo layout
  pack_weights<<<(FIN * HDIM + TB - 1) / TB, TB, 0, stream>>>(W0, pW0h, pW0l, FIN, HDIM);
  for (int i = 0; i < NLAYER; ++i) {
    size_t o = (size_t)i * HDIM * HDIM;
    pack_weights<<<(HDIM * HDIM + TB - 1) / TB, TB, 0, stream>>>(
        Ws + o, pWsh + o, pWsl + o, HDIM, HDIM);
  }
  pack_weights<<<(HDIM * FOUT + TB - 1) / TB, TB, 0, stream>>>(Wc, pWch, pWcl, HDIM, FOUT);

  // 3) x0 = relu(x @ W0 + b0)
  launch_gemm(x, pW0h, pW0l, x0, nullptr, b0, 0.0f, 1.0f, 1, NNODES, FIN, HDIM);

  // 4) GCNII layers
  const size_t nElem4 = (size_t)NNODES * (HDIM / 4);
  const int blk4 = (int)((nElem4 + TB - 1) / TB);
  const int spmmBlocks = (NEDGES + 3) / 4;     // 4 edges per 256-thread block
  const float* h_in = x0;
  for (int i = 0; i < NLAYER; ++i) {
    float beta = logf(THETA_ / (float)(i + 1) + 1.0f);
    float* h_out = (i == NLAYER - 1) ? outEmb : hA;

    agg_init<<<blk4, TB, 0, stream>>>(dinv, h_in, agg);
    spmm_kernel<<<spmmBlocks, TB, 0, stream>>>(srcE, dstE, dinv, h_in, agg, NEDGES);
    z_kernel<<<blk4, TB, 0, stream>>>(agg, x0, zb);

    size_t o = (size_t)i * HDIM * HDIM;
    // h = relu((1-beta)*z + beta*(z @ Ws[i]))
    launch_gemm(zb, pWsh + o, pWsl + o, h_out, zb, nullptr,
                1.0f - beta, beta, 1, NNODES, HDIM, HDIM);
    h_in = h_out;
  }

  // 5) logits = emb @ Wc + bc
  launch_gemm(outEmb, pWch, pWcl, outLogits, nullptr, bc,
              0.0f, 1.0f, 0, NNODES, HDIM, FOUT);

  // 6) softmax + argmax (one wave per row)
  softmax_argmax<<<(NNODES * 32 + TB - 1) / TB, TB, 0, stream>>>(outLogits, outSoft,
                                                                 outHard, NNODES);
}
